// MultiheadAttention_91122026152875
// MI455X (gfx1250) — matrix operand order repaired and checked by reading, not compiled, not run on hardware
//
#include <hip/hip_runtime.h>
#include <hip/hip_bf16.h>
#include <math.h>

typedef __attribute__((ext_vector_type(16))) __bf16 v16bf;
typedef __attribute__((ext_vector_type(8)))  float  v8f;

union Frag16 { v16bf v; uint4 u[2]; };

#define LSTR 48  // LDS row stride in bf16 elements (96B = 6x16B: aligned, staggers banks)

// async global->LDS 16B copy (ASYNCcnt-tracked); lds_addr = low 32 bits of the
// generic pointer into __shared__ (== LDS byte offset).
__device__ __forceinline__ void async_b128(unsigned int lds_addr, const void* gptr) {
  asm volatile("global_load_async_to_lds_b128 %0, %1, off"
               :: "v"(lds_addr), "v"((unsigned long long)(uintptr_t)gptr) : "memory");
}
template <int N>
__device__ __forceinline__ void wait_async() {
  asm volatile("s_wait_asynccnt %0" :: "i"(N) : "memory");
}

// ---------------------------------------------------------------- cast f32->bf16
__global__ __launch_bounds__(256) void cast_f32_bf16(const float* __restrict__ in,
                                                     __bf16* __restrict__ out, int n4) {
  int i = blockIdx.x * 256 + threadIdx.x;
  if (i >= n4) return;
  float4 f = ((const float4*)in)[i];
  union { __bf16 h[4]; uint2 u; } p;
  p.h[0] = (__bf16)f.x; p.h[1] = (__bf16)f.y; p.h[2] = (__bf16)f.z; p.h[3] = (__bf16)f.w;
  ((uint2*)out)[i] = p.u;
}

// ---------------------------------------------------------------- transpose + cast
// in f32 [K][N] -> out bf16 [N][K]; 64x64 tiles; grid (N/64, K/64)
__global__ __launch_bounds__(256) void transpose_cast(const float* __restrict__ in,
                                                      __bf16* __restrict__ out,
                                                      int K, int N) {
  __shared__ __align__(16) __bf16 t[64][72];   // 144B stride (9x16B)
  const int k0 = blockIdx.y * 64, n0 = blockIdx.x * 64;
  const int r = threadIdx.x >> 2, c4 = (threadIdx.x & 3) * 16;
  const float* ip = in + (size_t)(k0 + r) * N + n0 + c4;
  for (int j = 0; j < 16; j += 4) {
    float4 f = *(const float4*)(ip + j);
    t[c4 + j + 0][r] = (__bf16)f.x;
    t[c4 + j + 1][r] = (__bf16)f.y;
    t[c4 + j + 2][r] = (__bf16)f.z;
    t[c4 + j + 3][r] = (__bf16)f.w;
  }
  __syncthreads();
  __bf16* op = out + (size_t)(n0 + r) * K + k0 + c4;
  *(uint4*)(op + 0) = *(uint4*)&t[r][c4 + 0];
  *(uint4*)(op + 8) = *(uint4*)&t[r][c4 + 8];
}

// ---------------------------------------------------------------- bf16 WMMA GEMM
// C[M][N] = A[M][K] * BT[N][K]^T, both bf16. block = 256 thr (8 waves),
// block tile 128x128, wave tile 32x64, K-step 32, double-buffered async staging.
__global__ __launch_bounds__(256) void gemm_bf16_wmma(const __bf16* __restrict__ A,
                                                      const __bf16* __restrict__ BT,
                                                      void* __restrict__ Cout,
                                                      int M, int N, int K, int c_f32) {
  (void)M;
  __shared__ __align__(16) __bf16 a_s[2][128 * LSTR];
  __shared__ __align__(16) __bf16 b_s[2][128 * LSTR];
  const int tid = threadIdx.x;
  const int w = tid >> 5, lane = tid & 31, lm = lane & 15, half = lane >> 4;
  const int wm = w >> 1, wn = w & 1;                 // 4 m-waves x 2 n-waves
  const int m0 = blockIdx.y * 128, n0 = blockIdx.x * 128;

  // per-thread staging: rows r and r+64, 16B chunk cc8 (all bases hoisted)
  const int r = tid >> 2, cc8 = (tid & 3) * 8;
  const __bf16* ag0 = A  + (size_t)(m0 + r) * K + cc8;
  const __bf16* ag1 = ag0 + (size_t)64 * K;
  const __bf16* bg0 = BT + (size_t)(n0 + r) * K + cc8;
  const __bf16* bg1 = bg0 + (size_t)64 * K;
  const unsigned loff = (unsigned)(r * LSTR + cc8) * 2;
  const unsigned rs64 = (unsigned)(64 * LSTR) * 2;
  const unsigned al0 = (unsigned)(uintptr_t)(void*)a_s[0] + loff;
  const unsigned al1 = (unsigned)(uintptr_t)(void*)a_s[1] + loff;
  const unsigned bl0 = (unsigned)(uintptr_t)(void*)b_s[0] + loff;
  const unsigned bl1 = (unsigned)(uintptr_t)(void*)b_s[1] + loff;

  v8f acc[2][4] = {};

  auto issue = [&](unsigned al, unsigned bl, int kks) {
    async_b128(al,        ag0 + kks);
    async_b128(al + rs64, ag1 + kks);
    async_b128(bl,        bg0 + kks);
    async_b128(bl + rs64, bg1 + kks);
  };
  auto compute = [&](const __bf16* as, const __bf16* bs) {
    Frag16 af[2];
    for (int mi = 0; mi < 2; ++mi) {
      const __bf16* arow = as + (wm * 32 + mi * 16 + lm) * LSTR;
      af[mi].u[0] = *(const uint4*)(arow + half * 8);
      af[mi].u[1] = *(const uint4*)(arow + 16 + half * 8);
    }
    for (int nt = 0; nt < 4; ++nt) {
      Frag16 bfr;
      const __bf16* brow = bs + (wn * 64 + nt * 16 + lm) * LSTR;
      bfr.u[0] = *(const uint4*)(brow + half * 8);
      bfr.u[1] = *(const uint4*)(brow + 16 + half * 8);
      for (int mi = 0; mi < 2; ++mi)
        acc[mi][nt] = __builtin_amdgcn_wmma_f32_16x16x32_bf16(false, af[mi].v, false, bfr.v,
                                                              (short)0, acc[mi][nt], false, false);
    }
  };

  // software pipeline: tiles are 32 wide; loop body covers two tiles (bufs 0,1)
  issue(al0, bl0, 0);
  for (int kk = 0; kk < K; kk += 64) {
    issue(al1, bl1, kk + 32);            // next tile streams behind compute
    wait_async<4>();                     // in-order ASYNCcnt: tile kk complete
    __syncthreads();
    compute(a_s[0], b_s[0]);
    __syncthreads();

    if (kk + 64 < K) { issue(al0, bl0, kk + 64); wait_async<4>(); }
    else             { wait_async<0>(); }
    __syncthreads();
    compute(a_s[1], b_s[1]);
    __syncthreads();
  }

  // C layout: VGPR v -> row v + half*8, col = lane&15
  for (int mi = 0; mi < 2; ++mi)
    for (int nt = 0; nt < 4; ++nt)
      for (int v = 0; v < 8; ++v) {
        int row = m0 + wm * 32 + mi * 16 + half * 8 + v;
        int col = n0 + wn * 64 + nt * 16 + lm;
        float val = acc[mi][nt][v];
        if (c_f32) ((float*)Cout)[(size_t)row * N + col] = val;
        else       ((__bf16*)Cout)[(size_t)row * N + col] = (__bf16)val;
      }
}

// ---------------------------------------------------------------- RoPE + scatter to heads
__global__ __launch_bounds__(256) void rope_scatter(const __bf16* __restrict__ qkv,
                                                    __bf16* __restrict__ Q,
                                                    __bf16* __restrict__ Kd,
                                                    __bf16* __restrict__ Vt) {
  int gid = blockIdx.x * 256 + threadIdx.x;   // B*L*H*64 = 4,194,304
  int j = gid & 63;
  int h = (gid >> 6) & 15;
  int l = (gid >> 10) & 2047;
  int b = (gid >> 21) & 1;

  const size_t base = ((size_t)b * 2048 + l) * 6144 + h * 128;
  float q0 = (float)qkv[base + j],        q1 = (float)qkv[base + j + 64];
  float k0 = (float)qkv[base + 2048 + j], k1 = (float)qkv[base + 2048 + j + 64];
  float v0 = (float)qkv[base + 4096 + j], v1 = (float)qkv[base + 4096 + j + 64];

  float inv_freq = exp2f(-(float)j * 0.20762050593045952f);  // 10000^(-j/64)
  float s, c;
  __sincosf((float)l * inv_freq, &s, &c);

  const float qs = 0.08838834764831845f;  // 1/sqrt(128) folded into Q
  float qr0 = (q0 * c - q1 * s) * qs;
  float qr1 = (q1 * c + q0 * s) * qs;
  float kr0 = k0 * c - k1 * s;
  float kr1 = k1 * c + k0 * s;

  size_t ob = ((size_t)(b * 16 + h) * 2048 + l) * 128;
  Q[ob + j]       = (__bf16)qr0;
  Q[ob + j + 64]  = (__bf16)qr1;
  Kd[ob + j]      = (__bf16)kr0;
  Kd[ob + j + 64] = (__bf16)kr1;

  size_t vb = (size_t)(b * 16 + h) * 128;
  Vt[(vb + j) * 2048 + l]      = (__bf16)v0;
  Vt[(vb + j + 64) * 2048 + l] = (__bf16)v1;
}

// ---------------------------------------------------------------- flash attention
// grid (L/64, H, B); block 128 (4 waves); wave owns 16 q rows; streams 32-key tiles.
__global__ __launch_bounds__(128) void flash_attn(const __bf16* __restrict__ Q,
                                                  const __bf16* __restrict__ Kd,
                                                  const __bf16* __restrict__ Vt,
                                                  __bf16* __restrict__ Y) {
  __shared__ __align__(16) __bf16 p_s[4 * 16 * 32];  // per-wave 16x32 P tile
  const int tid = threadIdx.x, w = tid >> 5, lane = tid & 31;
  const int lm = lane & 15, half = lane >> 4;
  const int b = blockIdx.z, h = blockIdx.y, qb = blockIdx.x;
  const int L = 2048, HD = 128;
  const size_t bh = (size_t)(b * 16 + h);
  const __bf16* Qb = Q + bh * L * HD;
  const __bf16* Kb = Kd + bh * L * HD;
  const __bf16* Vb = Vt + bh * HD * L;
  const int q0 = qb * 64 + w * 16;

  Frag16 qf[4];
  for (int ks = 0; ks < 4; ++ks) {
    const __bf16* qp = Qb + (size_t)(q0 + lm) * HD + ks * 32;
    qf[ks].u[0] = *(const uint4*)(qp + half * 8);
    qf[ks].u[1] = *(const uint4*)(qp + 16 + half * 8);
  }

  v8f o[8] = {};
  float m_r[8], l_r[8];
  for (int i = 0; i < 8; ++i) { m_r[i] = -1e30f; l_r[i] = 0.0f; }

  __bf16* pw = p_s + w * 512;

  for (int kt = 0; kt < 64; ++kt) {
    if (kt + 1 < 64) {  // prefetch next K/V tiles (global_prefetch_b8)
      __builtin_prefetch(Kb + (size_t)((kt + 1) * 32 + lm) * HD, 0, 1);
      __builtin_prefetch(Vb + (size_t)lm * L + (kt + 1) * 32, 0, 1);
    }
    v8f S0 = {}, S1 = {};
    for (int ks = 0; ks < 4; ++ks) {
      Frag16 k0f, k1f;
      const __bf16* kp = Kb + (size_t)(kt * 32 + lm) * HD + ks * 32;
      k0f.u[0] = *(const uint4*)(kp + half * 8);       k0f.u[1] = *(const uint4*)(kp + 16 + half * 8);
      const __bf16* kp1 = kp + 16 * HD;
      k1f.u[0] = *(const uint4*)(kp1 + half * 8);      k1f.u[1] = *(const uint4*)(kp1 + 16 + half * 8);
      S0 = __builtin_amdgcn_wmma_f32_16x16x32_bf16(false, qf[ks].v, false, k0f.v,
                                                   (short)0, S0, false, false);
      S1 = __builtin_amdgcn_wmma_f32_16x16x32_bf16(false, qf[ks].v, false, k1f.v,
                                                   (short)0, S1, false, false);
    }

    float alpha[8];
    for (int v = 0; v < 8; ++v) {
      float mx = fmaxf(S0[v], S1[v]);
      mx = fmaxf(mx, __shfl_xor(mx, 1, 32));
      mx = fmaxf(mx, __shfl_xor(mx, 2, 32));
      mx = fmaxf(mx, __shfl_xor(mx, 4, 32));
      mx = fmaxf(mx, __shfl_xor(mx, 8, 32));
      float mn = fmaxf(m_r[v], mx);
      float a  = __expf(m_r[v] - mn);
      float p0 = __expf(S0[v] - mn);
      float p1 = __expf(S1[v] - mn);
      float rr = p0 + p1;
      rr += __shfl_xor(rr, 1, 32);
      rr += __shfl_xor(rr, 2, 32);
      rr += __shfl_xor(rr, 4, 32);
      rr += __shfl_xor(rr, 8, 32);
      l_r[v] = l_r[v] * a + rr;
      m_r[v] = mn;
      alpha[v] = a;
      __bf16* pr = pw + (half * 8 + v) * 32;   // wave-private; DS in-order per wave
      pr[lm]      = (__bf16)p0;
      pr[lm + 16] = (__bf16)p1;
    }
    for (int nt = 0; nt < 8; ++nt)
      for (int v = 0; v < 8; ++v) o[nt][v] = o[nt][v] * alpha[v];

    Frag16 pf;
    const __bf16* prow = pw + lm * 32;
    pf.u[0] = *(const uint4*)(prow + half * 8);
    pf.u[1] = *(const uint4*)(prow + 16 + half * 8);

    for (int nt = 0; nt < 8; ++nt) {
      Frag16 vf;
      const __bf16* vp = Vb + (size_t)(nt * 16 + lm) * L + kt * 32;
      vf.u[0] = *(const uint4*)(vp + half * 8);
      vf.u[1] = *(const uint4*)(vp + 16 + half * 8);
      o[nt] = __builtin_amdgcn_wmma_f32_16x16x32_bf16(false, pf.v, false, vf.v,
                                                      (short)0, o[nt], false, false);
    }
  }

  for (int v = 0; v < 8; ++v) {
    float inv = 1.0f / l_r[v];
    int row = q0 + half * 8 + v;
    for (int nt = 0; nt < 8; ++nt) {
      Y[((size_t)b * L + row) * 2048 + h * HD + nt * 16 + lm] = (__bf16)(o[nt][v] * inv);
    }
  }
}

// ---------------------------------------------------------------- launch
extern "C" void kernel_launch(void* const* d_in, const int* in_sizes, int n_in,
                              void* d_out, int out_size, void* d_ws, size_t ws_size,
                              hipStream_t stream) {
  (void)in_sizes; (void)n_in; (void)out_size; (void)ws_size;
  const float* x     = (const float*)d_in[0];  // [2][2048][2048]
  const float* w_qkv = (const float*)d_in[1];  // [2048][6144]
  const float* w_out = (const float*)d_in[2];  // [2048][2048]
  float* out = (float*)d_out;                  // [2][2048][2048]

  // workspace layout with lifetime overlap (peak 120 MB):
  //   [0,16M):    xb (phase 1-2)  then Qh (phase 3-4)
  //   [16M,40M):  wqT (phase 1-2) then Kh@16M, Vt@32M (phase 3-4)
  //   [48M,96M):  qkv
  //   [96M,104M): woT
  //   [104M,120M): Yb
  char* ws = (char*)d_ws;
  const size_t MB = 1024 * 1024;
  __bf16* xb  = (__bf16*)(ws);
  __bf16* wqT = (__bf16*)(ws + 16 * MB);
  __bf16* Qh  = (__bf16*)(ws);
  __bf16* Kh  = (__bf16*)(ws + 16 * MB);
  __bf16* Vt  = (__bf16*)(ws + 32 * MB);
  __bf16* qkv = (__bf16*)(ws + 48 * MB);
  __bf16* woT = (__bf16*)(ws + 96 * MB);
  __bf16* Yb  = (__bf16*)(ws + 104 * MB);

  // 1) casts / transposes (one-shot, bandwidth-light)
  cast_f32_bf16<<<8192, 256, 0, stream>>>(x, xb, 2097152);
  transpose_cast<<<dim3(96, 32), 256, 0, stream>>>(w_qkv, wqT, 2048, 6144);
  transpose_cast<<<dim3(32, 32), 256, 0, stream>>>(w_out, woT, 2048, 2048);

  // 2) QKV projection: [4096,2048] @ [2048,6144] -> bf16
  gemm_bf16_wmma<<<dim3(48, 32), 256, 0, stream>>>(xb, wqT, (void*)qkv,
                                                   4096, 6144, 2048, 0);

  // 3) RoPE + per-head scatter
  rope_scatter<<<16384, 256, 0, stream>>>(qkv, Qh, Kh, Vt);

  // 4) flash attention
  flash_attn<<<dim3(32, 16, 2), 128, 0, stream>>>(Qh, Kh, Vt, Yb);

  // 5) output projection -> f32 out
  gemm_bf16_wmma<<<dim3(16, 32), 256, 0, stream>>>(Yb, woT, (void*)out,
                                                   4096, 2048, 2048, 1);
}
